// VotePoseProposal_10539849744923
// MI455X (gfx1250) — matrix operand order repaired and checked by reading, not compiled, not run on hardware
//
#include <hip/hip_runtime.h>
#include <hip/hip_bf16.h>

typedef _Float16 v16h __attribute__((ext_vector_type(16)));
typedef _Float16 v8h  __attribute__((ext_vector_type(8)));
typedef float    v8f  __attribute__((ext_vector_type(8)));
typedef unsigned int v4u __attribute__((ext_vector_type(4)));
typedef int          v4i __attribute__((ext_vector_type(4)));
typedef int          v8i __attribute__((ext_vector_type(8)));

#define B_   8
#define K_   4096
#define C_   256
#define NP_  1024
#define NS_  16
#define R_   0.3f
#define R2_  0.09f
#define EPS_ 1e-5f

// ---------------- workspace layout (bytes) ----------------
#define O_FEATSH  0u                       // 8*4096*256 half  = 16,777,216
#define O_NEWXYZ  (16777216u)              // 8*1024*3  f32   = 98,304
#define O_IDX     (O_NEWXYZ + 98304u)      // 8*1024*16 i32   = 524,288
#define O_F       (O_IDX + 524288u)        // 8192*128  half  = 2,097,152
#define O_W0      (O_F + 2097152u)         // 9*8*512 half    = 73,728
#define O_W1      (O_W0 + 73728u)          // 4*8*512 half    = 32,768
#define O_W2      (O_W1 + 32768u)
#define O_HW1     (O_W2 + 32768u)
#define O_HW2     (O_HW1 + 32768u)
#define O_HW3     (O_HW2 + 32768u)         // 4*1*512 half    = 4,096
#define O_SCALES  (O_HW3 + 4096u)          // 5*128 f32
#define O_BIASES  (O_SCALES + 2560u)       // 5*128 f32
#define O_BIAS3   (O_BIASES + 2560u)       // 16 f32

// ---------------- WMMA helpers ----------------
__device__ __forceinline__ v8f wmma_f16(v16h a, v16h b, v8f c) {
  return __builtin_amdgcn_wmma_f32_16x16x32_f16(false, a, false, b, (short)0, c,
                                                false, false);
}

__device__ __forceinline__ v16h mk16(v8h lo, v8h hi) {
  v16h r;
#pragma unroll
  for (int i = 0; i < 8; ++i) { r[i] = lo[i]; r[i + 8] = hi[i]; }
  return r;
}

// A-matrix fragment (16-bit, 16xK row-major buffer, chunk kc of 32 K-values).
__device__ __forceinline__ v16h load_afrag(const _Float16* buf, int stride,
                                           int kc, int lane) {
  const int m   = lane & 15;
  const int off = (lane >> 4) * 8;
  const _Float16* row = buf + m * stride + kc * 32 + off;
  return mk16(*(const v8h*)row, *(const v8h*)(row + 16));
}

// B fragment: pre-packed per-lane 16 contiguous halves
__device__ __forceinline__ v16h load_bfrag(const _Float16* frag, int lane) {
  const _Float16* p = frag + lane * 16;
  return mk16(*(const v8h*)p, *(const v8h*)(p + 8));
}

// ---------------- weight prep ----------------
__global__ void prep_scales(const float* __restrict__ sa_b,
                            const float* __restrict__ sa_g,
                            const float* __restrict__ sa_be,
                            const float* __restrict__ sa_m,
                            const float* __restrict__ sa_v,
                            const float* __restrict__ h_b1,
                            const float* __restrict__ h_b2,
                            const float* __restrict__ h_b3,
                            const float* __restrict__ h_g,
                            const float* __restrict__ h_be,
                            const float* __restrict__ h_m,
                            const float* __restrict__ h_v,
                            float* __restrict__ scales,
                            float* __restrict__ biases,
                            float* __restrict__ bias3) {
  int o = threadIdx.x;
  if (o < 128) {
    for (int i = 0; i < 3; ++i) {
      float s = sa_g[i * 128 + o] * rsqrtf(sa_v[i * 128 + o] + EPS_);
      scales[i * 128 + o] = s;
      biases[i * 128 + o] =
          (sa_b[i * 128 + o] - sa_m[i * 128 + o]) * s + sa_be[i * 128 + o];
    }
    for (int j = 0; j < 2; ++j) {
      const float* hb = j ? h_b2 : h_b1;
      float s = h_g[j * 128 + o] * rsqrtf(h_v[j * 128 + o] + EPS_);
      scales[(3 + j) * 128 + o] = s;
      biases[(3 + j) * 128 + o] =
          (hb[o] - h_m[j * 128 + o]) * s + h_be[j * 128 + o];
    }
  }
  if (o < 16) bias3[o] = (o < 5) ? h_b3[o] : 0.0f;
}

// Pack W (Nv x Kld, row-major) into B fragments.
// out index = ((kc*NT+nt)*32 + lane)*16 + h ; K = kc*32 + (h>=8)*16 + (lane>=16)*8 + (h&7)
// xyzmode: A-row layout is [feat(256) | xyz(3) | pad], source cols remapped.
__global__ void pack_frags(const float* __restrict__ W,
                           const float* __restrict__ scale,
                           _Float16* __restrict__ out, int Nv, int Kld, int Kv,
                           int KC, int NT, int xyzmode) {
  int idx = blockIdx.x * blockDim.x + threadIdx.x;
  int total = KC * NT * 512;
  if (idx >= total) return;
  int h    = idx & 15;
  int lane = (idx >> 4) & 31;
  int tile = idx >> 9;
  int nt   = tile % NT;
  int kc   = tile / NT;
  int Kg   = kc * 32 + ((h >> 3) << 4) + ((lane >> 4) << 3) + (h & 7);
  int N    = nt * 16 + (lane & 15);
  int srcK;
  if (xyzmode) {
    if (Kg < 256)      srcK = Kg + 3;       // feature channels
    else if (Kg < 259) srcK = Kg - 256;     // xyz channels
    else               srcK = -1;           // pad
  } else {
    srcK = (Kg < Kv) ? Kg : -1;
  }
  float v = 0.0f;
  if (srcK >= 0 && N < Nv) {
    v = W[N * Kld + srcK];
    if (scale) v *= scale[N];
  }
  out[idx] = (_Float16)v;
}

// ---------------- feature transpose (B,C,K)f32 -> (B,K,C)f16 ----------------
__global__ __launch_bounds__(256) void transpose_feats(
    const float* __restrict__ f, _Float16* __restrict__ o) {
  __shared__ float tile[32][33];
  int b = blockIdx.z, c0 = blockIdx.y * 32, k0 = blockIdx.x * 32;
  int tx = threadIdx.x, ty = threadIdx.y;
  for (int i = ty; i < 32; i += 8)
    tile[i][tx] = f[((size_t)b * C_ + c0 + i) * K_ + k0 + tx];
  __syncthreads();
  for (int i = ty; i < 32; i += 8)
    o[((size_t)b * K_ + k0 + i) * C_ + c0 + tx] = (_Float16)tile[tx][i];
}

// ---------------- farthest point sampling ----------------
__global__ __launch_bounds__(256) void fps_kernel(const float* __restrict__ xyz,
                                                  float* __restrict__ newxyz) {
  const int b = blockIdx.x, tid = threadIdx.x;
  const float* xb = xyz + (size_t)b * K_ * 3;
  float px[16], py[16], pz[16], dm[16];
#pragma unroll
  for (int t = 0; t < 16; ++t) {
    int k = t * 256 + tid;
    px[t] = xb[k * 3 + 0];
    py[t] = xb[k * 3 + 1];
    pz[t] = xb[k * 3 + 2];
    dm[t] = 1e10f;
  }
  __shared__ float s_bv[2][8];
  __shared__ int   s_bi[2][8];
  float lx = xb[0], ly = xb[1], lz = xb[2];
  if (tid == 0) {
    float* o = newxyz + (size_t)b * NP_ * 3;
    o[0] = lx; o[1] = ly; o[2] = lz;
  }
  for (int j = 1; j < NP_; ++j) {
    float bv = -1.0f; int bi = 0;
#pragma unroll
    for (int t = 0; t < 16; ++t) {
      float dx = px[t] - lx, dy = py[t] - ly, dz = pz[t] - lz;
      float d = dx * dx + dy * dy + dz * dz;
      dm[t] = fminf(dm[t], d);
      int k = t * 256 + tid;
      if (dm[t] > bv || (dm[t] == bv && k < bi)) { bv = dm[t]; bi = k; }
    }
#pragma unroll
    for (int off = 16; off >= 1; off >>= 1) {
      float ov = __shfl_down(bv, off, 32);
      int   oi = __shfl_down(bi, off, 32);
      if (ov > bv || (ov == bv && oi < bi)) { bv = ov; bi = oi; }
    }
    const int buf = j & 1;
    if ((tid & 31) == 0) { s_bv[buf][tid >> 5] = bv; s_bi[buf][tid >> 5] = bi; }
    __syncthreads();
    bv = s_bv[buf][0]; bi = s_bi[buf][0];
#pragma unroll
    for (int w = 1; w < 8; ++w) {
      float ov = s_bv[buf][w]; int oi = s_bi[buf][w];
      if (ov > bv || (ov == bv && oi < bi)) { bv = ov; bi = oi; }
    }
    lx = xb[bi * 3 + 0]; ly = xb[bi * 3 + 1]; lz = xb[bi * 3 + 2];
    if (tid == 0) {
      float* o = newxyz + ((size_t)b * NP_ + j) * 3;
      o[0] = lx; o[1] = ly; o[2] = lz;
    }
  }
}

// ---------------- ball query: first NS indices with d2 < R2 ----------------
__global__ __launch_bounds__(256) void ballq_kernel(
    const float* __restrict__ xyz, const float* __restrict__ newxyz,
    int* __restrict__ idxb) {
  __shared__ float sx[K_], sy[K_], sz[K_];
  const int b = blockIdx.x >> 2;
  const int p = ((blockIdx.x & 3) << 8) + threadIdx.x;
  const float* xb = xyz + (size_t)b * K_ * 3;
  for (int i = threadIdx.x; i < K_; i += 256) {
    sx[i] = xb[i * 3 + 0]; sy[i] = xb[i * 3 + 1]; sz[i] = xb[i * 3 + 2];
  }
  __syncthreads();
  const int gp = b * NP_ + p;
  float cx = newxyz[gp * 3 + 0], cy = newxyz[gp * 3 + 1],
        cz = newxyz[gp * 3 + 2];
  int* dst = idxb + (size_t)gp * NS_;
  int cnt = 0, first = 0;
  for (int k = 0; k < K_; ++k) {
    float dx = sx[k] - cx, dy = sy[k] - cy, dz = sz[k] - cz;
    float d = dx * dx + dy * dy + dz * dz;
    if (d < R2_) {
      if (cnt == 0) first = k;
      dst[cnt] = k;
      if (++cnt == NS_) break;
    }
  }
  for (int t = cnt; t < NS_; ++t) dst[t] = first;
}

// ---------------- fused SA MLP (3 layers) + max over samples ----------------
// Neighbor feature rows are gathered by the Tensor Data Mover in gather mode:
// one tensor_load_to_lds per wave fetches 16 rows x 512B into sFeat.
__global__ __launch_bounds__(128) void sa_kernel(
    const float* __restrict__ xyz, const _Float16* __restrict__ feats,
    const float* __restrict__ newxyz, const int* __restrict__ idxb,
    const _Float16* __restrict__ w0, const _Float16* __restrict__ w1,
    const _Float16* __restrict__ w2, const float* __restrict__ bias0,
    const float* __restrict__ bias1, const float* __restrict__ bias2,
    _Float16* __restrict__ fout) {
  __shared__ __align__(16) _Float16 sFeat[4][16 * 256];  // TDM gather dest
  __shared__ __align__(16) _Float16 sXyz[4][16 * 32];    // g_xyz + zero pad
  __shared__ __align__(16) _Float16 sH[4][16 * 128];
  const int wave = threadIdx.x >> 5;
  const int lane = threadIdx.x & 31;
  const int pt   = blockIdx.x * 4 + wave;     // 0..8191
  const int b    = pt >> 10;
  const float* xb = xyz + (size_t)b * K_ * 3;

  // per-lane neighbor id (lanes 16..31 mirror 0..15)
  int id = idxb[(size_t)pt * NS_ + (lane & 15)];
  if (id < 0) id = 0;
  if (id > K_ - 1) id = K_ - 1;

  // pack the 16 row indices into 8 wave-uniform dwords (D# groups 2/3)
  int dw[8];
#pragma unroll
  for (int i = 0; i < 8; ++i) {
    int lo = __builtin_amdgcn_readlane(id, 2 * i);
    int hi = __builtin_amdgcn_readlane(id, 2 * i + 1);
    dw[i] = (lo & 0xffff) | (hi << 16);
  }

  // Tensor DMA descriptor: gather 16 rows of 256 half elements from
  // feats[b] (4096 x 256, row stride 256 elements) into sFeat[wave].
  unsigned long long ga =
      (unsigned long long)(uintptr_t)(feats + (size_t)b * K_ * C_);
  unsigned int ldsa = (unsigned int)(uintptr_t)&sFeat[wave][0];
  v4u g0 = {0x80000001u,                        // count=1, gather_mode=1, 16b idx
            ldsa,                               // lds_addr
            (unsigned int)ga,                   // global_addr[31:0]
            (unsigned int)((ga >> 32) & 0x01FFFFFFull) | 0x80000000u};  // +type=2
  v8i g1 = {0x00010000,            // data_size=2B
            (int)(256u << 16),     // tensor_dim0 = 256
            (int)(4096u << 16),    // tensor_dim1 = 4096
            (int)(256u << 16),     // tile_dim0 = 256
            16,                    // tile_dim1 = #indices = 16
            256,                   // tensor_dim0_stride = 256 elements
            0, 0};
  v4i g2 = {dw[0], dw[1], dw[2], dw[3]};
  v4i g3 = {dw[4], dw[5], dw[6], dw[7]};
#if defined(__clang_major__) && __clang_major__ >= 23
  v8i gz = {0, 0, 0, 0, 0, 0, 0, 0};
  __builtin_amdgcn_tensor_load_to_lds(g0, g1, g2, g3, gz, 0);
#else
  __builtin_amdgcn_tensor_load_to_lds(g0, g1, g2, g3, 0);
#endif

  // g_xyz rows: [gx gy gz 0...0] (32 halves) = K-chunk 8 of layer 0
  if (lane < 16) {
    float cx = newxyz[pt * 3 + 0], cy = newxyz[pt * 3 + 1],
          cz = newxyz[pt * 3 + 2];
    _Float16* row = &sXyz[wave][lane * 32];
    row[0] = (_Float16)((xb[id * 3 + 0] - cx) * (1.0f / R_));
    row[1] = (_Float16)((xb[id * 3 + 1] - cy) * (1.0f / R_));
    row[2] = (_Float16)((xb[id * 3 + 2] - cz) * (1.0f / R_));
#pragma unroll
    for (int h = 3; h < 32; ++h) row[h] = (_Float16)0.0f;
  }
  __builtin_amdgcn_s_wait_tensorcnt(0);
  __syncthreads();

  // Layer 0: (16x288) x (288x128); K-chunks 0..7 from sFeat, chunk 8 from sXyz
  for (int nt = 0; nt < 8; ++nt) {
    v8f acc = {0, 0, 0, 0, 0, 0, 0, 0};
#pragma unroll
    for (int kc = 0; kc < 8; ++kc) {
      v16h a  = load_afrag(&sFeat[wave][0], 256, kc, lane);
      v16h bm = load_bfrag(w0 + (size_t)((kc * 8 + nt) * 32) * 16, lane);
      acc = wmma_f16(a, bm, acc);
    }
    {
      v16h a  = load_afrag(&sXyz[wave][0], 32, 0, lane);
      v16h bm = load_bfrag(w0 + (size_t)((8 * 8 + nt) * 32) * 16, lane);
      acc = wmma_f16(a, bm, acc);
    }
    const int n  = nt * 16 + (lane & 15);
    const int mb = (lane >> 4) * 8;
    const float bs = bias0[n];
#pragma unroll
    for (int r = 0; r < 8; ++r) {
      float v = acc[r] + bs;
      v = v > 0.0f ? v : 0.0f;
      sH[wave][(mb + r) * 128 + n] = (_Float16)v;
    }
  }
  __syncthreads();

  // Layer 1: (16x128) x (128x128) -> h1 stored into sFeat region
  for (int nt = 0; nt < 8; ++nt) {
    v8f acc = {0, 0, 0, 0, 0, 0, 0, 0};
#pragma unroll
    for (int kc = 0; kc < 4; ++kc) {
      v16h a  = load_afrag(&sH[wave][0], 128, kc, lane);
      v16h bm = load_bfrag(w1 + (size_t)((kc * 8 + nt) * 32) * 16, lane);
      acc = wmma_f16(a, bm, acc);
    }
    const int n  = nt * 16 + (lane & 15);
    const int mb = (lane >> 4) * 8;
    const float bs = bias1[n];
#pragma unroll
    for (int r = 0; r < 8; ++r) {
      float v = acc[r] + bs;
      v = v > 0.0f ? v : 0.0f;
      sFeat[wave][(mb + r) * 128 + n] = (_Float16)v;
    }
  }
  __syncthreads();

  // Layer 2 + max over the 16 samples (relu is monotone -> max after relu)
  for (int nt = 0; nt < 8; ++nt) {
    v8f acc = {0, 0, 0, 0, 0, 0, 0, 0};
#pragma unroll
    for (int kc = 0; kc < 4; ++kc) {
      v16h a  = load_afrag(&sFeat[wave][0], 128, kc, lane);
      v16h bm = load_bfrag(w2 + (size_t)((kc * 8 + nt) * 32) * 16, lane);
      acc = wmma_f16(a, bm, acc);
    }
    const int n = nt * 16 + (lane & 15);
    const float bs = bias2[n];
    float mx = 0.0f;  // relu output >= 0
#pragma unroll
    for (int r = 0; r < 8; ++r) {
      float v = acc[r] + bs;
      v = v > 0.0f ? v : 0.0f;
      mx = fmaxf(mx, v);
    }
    float ox = __shfl_xor(mx, 16, 32);  // combine row halves (M 0-7 vs 8-15)
    mx = fmaxf(mx, ox);
    if (lane < 16) fout[(size_t)pt * 128 + n] = (_Float16)mx;
  }
}

// ---------------- head MLP: 16 points per wave as M dimension ----------------
__global__ __launch_bounds__(128) void head_kernel(
    const _Float16* __restrict__ fbuf, const _Float16* __restrict__ hw1,
    const _Float16* __restrict__ hw2, const _Float16* __restrict__ hw3,
    const float* __restrict__ biasH1, const float* __restrict__ biasH2,
    const float* __restrict__ bias3, const float* __restrict__ newxyz,
    float* __restrict__ out) {
  __shared__ __align__(16) _Float16 s0[4][16 * 128];
  __shared__ __align__(16) _Float16 s1[4][16 * 128];
  const int wave = threadIdx.x >> 5;
  const int lane = threadIdx.x & 31;
  const int p0 = (blockIdx.x * 4 + wave) * 16;  // 512 tiles total
  const int m  = lane & 15;
  const int off = (lane >> 4) * 8;

  // H1: A rows straight from global f (rows are 256B contiguous)
  for (int nt = 0; nt < 8; ++nt) {
    v8f acc = {0, 0, 0, 0, 0, 0, 0, 0};
#pragma unroll
    for (int kc = 0; kc < 4; ++kc) {
      const _Float16* row = fbuf + (size_t)(p0 + m) * 128 + kc * 32 + off;
      v16h a  = mk16(*(const v8h*)row, *(const v8h*)(row + 16));
      v16h bm = load_bfrag(hw1 + (size_t)((kc * 8 + nt) * 32) * 16, lane);
      acc = wmma_f16(a, bm, acc);
    }
    const int n  = nt * 16 + (lane & 15);
    const int mb = (lane >> 4) * 8;
    const float bs = biasH1[n];
#pragma unroll
    for (int r = 0; r < 8; ++r) {
      float v = acc[r] + bs;
      v = v > 0.0f ? v : 0.0f;
      s0[wave][(mb + r) * 128 + n] = (_Float16)v;
    }
  }
  __syncthreads();

  // H2
  for (int nt = 0; nt < 8; ++nt) {
    v8f acc = {0, 0, 0, 0, 0, 0, 0, 0};
#pragma unroll
    for (int kc = 0; kc < 4; ++kc) {
      v16h a  = load_afrag(&s0[wave][0], 128, kc, lane);
      v16h bm = load_bfrag(hw2 + (size_t)((kc * 8 + nt) * 32) * 16, lane);
      acc = wmma_f16(a, bm, acc);
    }
    const int n  = nt * 16 + (lane & 15);
    const int mb = (lane >> 4) * 8;
    const float bs = biasH2[n];
#pragma unroll
    for (int r = 0; r < 8; ++r) {
      float v = acc[r] + bs;
      v = v > 0.0f ? v : 0.0f;
      s1[wave][(mb + r) * 128 + n] = (_Float16)v;
    }
  }
  __syncthreads();

  // H3: single N-tile (5 valid cols), then write output (+center offset)
  {
    v8f acc = {0, 0, 0, 0, 0, 0, 0, 0};
#pragma unroll
    for (int kc = 0; kc < 4; ++kc) {
      v16h a  = load_afrag(&s1[wave][0], 128, kc, lane);
      v16h bm = load_bfrag(hw3 + (size_t)(kc * 32) * 16, lane);
      acc = wmma_f16(a, bm, acc);
    }
    const int n  = lane & 15;
    const int mb = (lane >> 4) * 8;
    if (n < 5) {
      const float bs = bias3[n];
#pragma unroll
      for (int r = 0; r < 8; ++r) {
        const int P = p0 + mb + r;
        float v = acc[r] + bs;
        if (n >= 2) v += newxyz[(size_t)P * 3 + (n - 2)];
        out[(size_t)P * 5 + n] = v;
      }
    }
  }
}

// ---------------- launcher ----------------
extern "C" void kernel_launch(void* const* d_in, const int* in_sizes, int n_in,
                              void* d_out, int out_size, void* d_ws,
                              size_t ws_size, hipStream_t stream) {
  (void)in_sizes; (void)n_in; (void)out_size; (void)ws_size;
  const float* xyz      = (const float*)d_in[0];
  const float* features = (const float*)d_in[1];
  const float* sa_w0 = (const float*)d_in[2];
  const float* sa_w1 = (const float*)d_in[3];
  const float* sa_w2 = (const float*)d_in[4];
  const float* sa_b  = (const float*)d_in[5];
  const float* sa_g  = (const float*)d_in[6];
  const float* sa_be = (const float*)d_in[7];
  const float* sa_m  = (const float*)d_in[8];
  const float* sa_v  = (const float*)d_in[9];
  const float* h_w1  = (const float*)d_in[10];
  const float* h_b1  = (const float*)d_in[11];
  const float* h_w2  = (const float*)d_in[12];
  const float* h_b2  = (const float*)d_in[13];
  const float* h_w3  = (const float*)d_in[14];
  const float* h_b3  = (const float*)d_in[15];
  const float* h_g   = (const float*)d_in[16];
  const float* h_be  = (const float*)d_in[17];
  const float* h_m   = (const float*)d_in[18];
  const float* h_v   = (const float*)d_in[19];
  float* out = (float*)d_out;
  char*  ws  = (char*)d_ws;

  _Float16* featsh = (_Float16*)(ws + O_FEATSH);
  float*    newxyz = (float*)(ws + O_NEWXYZ);
  int*      idxb   = (int*)(ws + O_IDX);
  _Float16* fbuf   = (_Float16*)(ws + O_F);
  _Float16* w0f    = (_Float16*)(ws + O_W0);
  _Float16* w1f    = (_Float16*)(ws + O_W1);
  _Float16* w2f    = (_Float16*)(ws + O_W2);
  _Float16* hw1f   = (_Float16*)(ws + O_HW1);
  _Float16* hw2f   = (_Float16*)(ws + O_HW2);
  _Float16* hw3f   = (_Float16*)(ws + O_HW3);
  float*    scales = (float*)(ws + O_SCALES);
  float*    biases = (float*)(ws + O_BIASES);
  float*    bias3  = (float*)(ws + O_BIAS3);

  prep_scales<<<1, 128, 0, stream>>>(sa_b, sa_g, sa_be, sa_m, sa_v, h_b1, h_b2,
                                     h_b3, h_g, h_be, h_m, h_v, scales, biases,
                                     bias3);
  pack_frags<<<144, 256, 0, stream>>>(sa_w0, scales + 0,   w0f, 128, 259, 259, 9, 8, 1);
  pack_frags<<<64,  256, 0, stream>>>(sa_w1, scales + 128, w1f, 128, 128, 128, 4, 8, 0);
  pack_frags<<<64,  256, 0, stream>>>(sa_w2, scales + 256, w2f, 128, 128, 128, 4, 8, 0);
  pack_frags<<<64,  256, 0, stream>>>(h_w1,  scales + 384, hw1f, 128, 128, 128, 4, 8, 0);
  pack_frags<<<64,  256, 0, stream>>>(h_w2,  scales + 512, hw2f, 128, 128, 128, 4, 8, 0);
  pack_frags<<<8,   256, 0, stream>>>(h_w3,  (const float*)nullptr, hw3f, 5, 128, 128, 4, 1, 0);

  transpose_feats<<<dim3(K_ / 32, C_ / 32, B_), dim3(32, 8), 0, stream>>>(
      features, featsh);
  fps_kernel<<<B_, 256, 0, stream>>>(xyz, newxyz);
  ballq_kernel<<<B_ * 4, 256, 0, stream>>>(xyz, newxyz, idxb);
  sa_kernel<<<(B_ * NP_) / 4, 128, 0, stream>>>(
      xyz, featsh, newxyz, idxb, w0f, w1f, w2f, biases + 0, biases + 128,
      biases + 256, fbuf);
  head_kernel<<<(B_ * NP_) / 16 / 4, 128, 0, stream>>>(
      fbuf, hw1f, hw2f, hw3f, biases + 384, biases + 512, bias3, newxyz, out);
}
